// Volume2XYZ_22806276342482
// MI455X (gfx1250) — matrix operand order repaired and checked by reading, not compiled, not run on hardware
//
#include <hip/hip_runtime.h>

typedef __attribute__((ext_vector_type(16))) _Float16 v16h;
typedef __attribute__((ext_vector_type(8)))  _Float16 v8h;
typedef __attribute__((ext_vector_type(8)))  float    v8f;

#define KPEAKS   131072
#define VOL_D    128
#define VOL_H    512
#define VOL_W    512
#define NVOX     (VOL_D * VOL_H * VOL_W)   // 33,554,432
#define THRESH   0.997f
#define PS_XY    0.1f
#define PS_Z     0.02f
#define ZMIN     (-2.0f)

#define KPAD     352                        // 343 rounded up to 11 x 32
#define WAVES    4                          // waves per centroid block

// ---------------------------------------------------------------------------
// Kernel 0: reset workspace (peak list + counter). Deterministic per call.
// ---------------------------------------------------------------------------
__global__ __launch_bounds__(256) void vp_init(float* __restrict__ vals,
                                               int* __restrict__ idxs,
                                               unsigned* __restrict__ cnt) {
    int i = blockIdx.x * 256 + threadIdx.x;
    vals[i] = 0.0f;
    idxs[i] = 0;
    if (i == 0) *cnt = 0u;
}

// ---------------------------------------------------------------------------
// Kernel 1: streaming peak detection. v is a peak iff v > THRESH and
// v >= every raw value in its 7x7x7 window (equivalent to the reference's
// thresholded max-pool test). 99.7% of lanes early-out after one load ->
// near pure stream of the 134 MB volume (~5.7us at 23.3 TB/s).
// ---------------------------------------------------------------------------
__global__ __launch_bounds__(256) void vp_detect(const float* __restrict__ vol,
                                                 float* __restrict__ vals,
                                                 int* __restrict__ idxs,
                                                 unsigned* __restrict__ cnt) {
    int i = blockIdx.x * 256 + threadIdx.x;
    __builtin_prefetch(vol + i + 32768, 0, 1);   // gfx1250 global_prefetch_b8
    float v = vol[i];
    if (v <= THRESH) return;

    int z = i >> 18;
    int y = (i >> 9) & 511;
    int x = i & 511;

    bool isPeak = true;
    for (int dz = -3; dz <= 3 && isPeak; ++dz) {
        int zz = z + dz;
        if ((unsigned)zz >= (unsigned)VOL_D) continue;
        for (int dy = -3; dy <= 3 && isPeak; ++dy) {
            int yy = y + dy;
            if ((unsigned)yy >= (unsigned)VOL_H) continue;
            const float* row = vol + ((zz << 18) + (yy << 9));
            for (int dx = -3; dx <= 3; ++dx) {
                int xx = x + dx;
                if ((unsigned)xx >= (unsigned)VOL_W) continue;
                if (row[xx] > v) { isPeak = false; break; }
            }
        }
    }
    if (isPeak) {
        unsigned p = atomicAdd(cnt, 1u);
        if (p < KPEAKS) { vals[p] = v; idxs[p] = i; }
    }
}

// ---------------------------------------------------------------------------
// Kernel 2: bitonic compare-exchange pass. Order: val descending, idx
// ascending tie-break (top_k semantics); zero-sentinel slots sink. Makes the
// final order deterministic regardless of atomic append order.
// ---------------------------------------------------------------------------
__global__ __launch_bounds__(256) void vp_bitonic(float* __restrict__ vals,
                                                  int* __restrict__ idxs,
                                                  unsigned j, unsigned k) {
    unsigned i = blockIdx.x * 256 + threadIdx.x;
    unsigned ixj = i ^ j;
    if (ixj <= i) return;
    float va = vals[i], vb = vals[ixj];
    int   ia = idxs[i], ib = idxs[ixj];
    bool a_first = (va > vb) || (va == vb && ia < ib);
    bool b_first = (vb > va) || (vb == va && ib < ia);
    bool doSwap = ((i & k) == 0) ? b_first : a_first;
    if (doSwap) {
        vals[i] = vb; vals[ixj] = va;
        idxs[i] = ib; idxs[ixj] = ia;
    }
}

// ---------------------------------------------------------------------------
// Kernel 3: per-peak 7x7x7 center-of-mass as a WMMA GEMM, LDS-staged.
//
//   D(16x16,f32) += A(16 peaks x 32k, f16) x B(32k x 16 cols, f16)
//   cols of B = [sum, dx, dy, dz] weights, 11 K-steps over 343 (padded 352).
//
// LDS layouts are linear-in-k, so per the CDNA5 wave32 fragment layouts every
// A and B fragment = 2 contiguous 8-half runs
// per lane -> pure ds_load_b128 in the hot loop, no index math.
// ---------------------------------------------------------------------------
__global__ __launch_bounds__(32 * WAVES) void
vp_centroid(const float* __restrict__ vol,
            const float* __restrict__ svals,
            const int* __restrict__ sidx,
            float* __restrict__ out,
            float* __restrict__ outValid) {
    __shared__ __align__(16) _Float16 wTab[16 * KPAD];           // 11 KB
    __shared__ __align__(16) _Float16 patch[WAVES][16 * KPAD];   // 45 KB
    __shared__ int   coordI[WAVES][16];
    __shared__ float dmat[WAVES][256];                           // 4 KB

    const int tid  = threadIdx.x;
    const int lane = tid & 31;
    const int wv   = tid >> 5;
    const int grp  = blockIdx.x * WAVES + wv;
    const int m    = lane & 15;          // peak row (A) / output col (B,D)
    const int g    = grp * 16 + m;
    const int hi   = (lane >= 16) ? 1 : 0;

    // ---- build the constant weight table [n][k] (whole block) ----
    for (int e = tid; e < 16 * KPAD; e += 32 * WAVES) {
        int n = e / KPAD, k = e % KPAD;
        float w = 0.0f;
        if (k < 343 && n < 4) {
            int dz = k / 49, r = k % 49, dy = r / 7, dx = r % 7;
            w = (n == 0) ? 1.0f
              : (n == 1) ? (float)(dx - 3)
              : (n == 2) ? (float)(dy - 3)
              :            (float)(dz - 3);
        }
        wTab[e] = (_Float16)w;
    }

    // ---- peak coordinates ----
    float pv   = svals[g];
    int   pidx = sidx[g];
    if (lane < 16) coordI[wv][lane] = pidx;
    __syncthreads();

    // ---- stage 16 patches as f16 into LDS: patch[wv][peak*KPAD + k] ----
    // 16 peaks x 49 (dz,dy)-rows = 784 row tasks, 7 clamped loads each.
    for (int t = 0; t < 25; ++t) {
        int id = t * 32 + lane;
        if (id < 784) {
            int pk = id / 49;
            int r  = id % 49;
            int ci = coordI[wv][pk];
            int cz = (ci >> 18)        + (r / 7) - 3;
            int cy = ((ci >> 9) & 511) + (r % 7) - 3;
            int cx = (ci & 511);
            bool rowok = ((unsigned)cz < (unsigned)VOL_D) &&
                         ((unsigned)cy < (unsigned)VOL_H);
            const float* rowp = vol + (((cz & 127) << 18) + ((cy & 511) << 9));
            _Float16* dst = &patch[wv][pk * KPAD + r * 7];
#pragma unroll
            for (int dx = 0; dx < 7; ++dx) {
                int xx = cx + dx - 3;
                bool ok = rowok && ((unsigned)xx < (unsigned)VOL_W);
                float v = rowp[xx & 511];          // clamped, always in-bounds
                dst[dx] = (_Float16)(ok ? v : 0.0f);
            }
        }
    }
    // zero the k = 343..351 pad of each peak
    if (lane < 16) {
#pragma unroll
        for (int k = 343; k < KPAD; ++k)
            patch[wv][lane * KPAD + k] = (_Float16)0.0f;
    }
    __syncthreads();

    // ---- WMMA main loop: fragments are contiguous LDS b128 loads ----
    const int aBase = m * KPAD;               // patch row for this lane
    const int bBase = m * KPAD;               // wTab column n == m == lane&15
    v8f acc = {};
#pragma unroll
    for (int kb = 0; kb < KPAD; kb += 32) {
        // A fragment: halves 0..7 <- k in [kb+8*hi, +8),
        //             halves 8..15 <- k in [kb+16+8*hi, +8)
        v8h a0 = *(const v8h*)&patch[wv][aBase + kb + 8 * hi];
        v8h a1 = *(const v8h*)&patch[wv][aBase + kb + 16 + 8 * hi];
        // B fragment: halves 0..7 <- k in [kb+8*hi, +8),
        //             halves 8..15 <- k in [kb+16+8*hi, +8)
        v8h b0 = *(const v8h*)&wTab[bBase + kb + 8 * hi];
        v8h b1 = *(const v8h*)&wTab[bBase + kb + 16 + 8 * hi];
        v16h A = __builtin_shufflevector(a0, a1, 0, 1, 2, 3, 4, 5, 6, 7,
                                                 8, 9, 10, 11, 12, 13, 14, 15);
        v16h B = __builtin_shufflevector(b0, b1, 0, 1, 2, 3, 4, 5, 6, 7,
                                                 8, 9, 10, 11, 12, 13, 14, 15);
        acc = __builtin_amdgcn_wmma_f32_16x16x32_f16(
                  false, A, false, B, (short)0, acc, false, false);
    }

    // ---- scatter D per C/D layout: element (M = v + 8*hi, N = lane&15) ----
#pragma unroll
    for (int v = 0; v < 8; ++v)
        dmat[wv][(v + 8 * hi) * 16 + m] = acc[v];
    __syncthreads();

    if (lane < 16) {
        float s0 = dmat[wv][lane * 16 + 0];
        float sx = dmat[wv][lane * 16 + 1];
        float sy = dmat[wv][lane * 16 + 2];
        float sz = dmat[wv][lane * 16 + 3];
        int px = pidx & 511, py = (pidx >> 9) & 511, pz = pidx >> 18;
        bool valid = pv > 0.0f;
        float inv = (s0 != 0.0f) ? (1.0f / s0) : 0.0f;
        float xr = ((float)px + sx * inv - (float)(VOL_W - 1) * 0.5f) * PS_XY;
        float yr = ((float)py + sy * inv - (float)(VOL_H - 1) * 0.5f) * PS_XY;
        float zr = ((float)pz + sz * inv + 0.5f) * PS_Z + ZMIN;
        float4 o;
        o.x = valid ? xr : 0.0f;
        o.y = valid ? yr : 0.0f;
        o.z = valid ? zr : 0.0f;
        o.w = pv;
        ((float4*)out)[g] = o;
        outValid[g] = valid ? 1.0f : 0.0f;
    }
}

// ---------------------------------------------------------------------------
extern "C" void kernel_launch(void* const* d_in, const int* in_sizes, int n_in,
                              void* d_out, int out_size, void* d_ws, size_t ws_size,
                              hipStream_t stream) {
    (void)in_sizes; (void)n_in; (void)out_size; (void)ws_size;

    const float* vol = (const float*)d_in[0];   // (1,1,128,512,512) f32

    float* out      = (float*)d_out;            // (K,4) f32
    float* outValid = out + (size_t)KPEAKS * 4; // (K,) as 0/1 floats

    char*     ws     = (char*)d_ws;
    float*    wsVals = (float*)ws;                               // K f32
    int*      wsIdx  = (int*)(ws + (size_t)KPEAKS * 4);          // K i32
    unsigned* wsCnt  = (unsigned*)(ws + (size_t)KPEAKS * 8);     // 1 u32

    vp_init<<<KPEAKS / 256, 256, 0, stream>>>(wsVals, wsIdx, wsCnt);

    vp_detect<<<NVOX / 256, 256, 0, stream>>>(vol, wsVals, wsIdx, wsCnt);

    for (unsigned k = 2; k <= (unsigned)KPEAKS; k <<= 1)
        for (unsigned j = k >> 1; j > 0; j >>= 1)
            vp_bitonic<<<KPEAKS / 256, 256, 0, stream>>>(wsVals, wsIdx, j, k);

    // 4 waves/block, 16 peaks/wave -> 2048 blocks cover K=131072 peaks
    vp_centroid<<<KPEAKS / (16 * WAVES), 32 * WAVES, 0, stream>>>(
        vol, wsVals, wsIdx, out, outValid);
}
